// DecoderSparse_80410377716044
// MI455X (gfx1250) — matrix operand order repaired and checked by reading, not compiled, not run on hardware
//
#include <hip/hip_runtime.h>
#include <hip/hip_bf16.h>
#include <stdint.h>

typedef __attribute__((ext_vector_type(16))) __bf16 v16bf;
typedef __attribute__((ext_vector_type(8)))  float  v8f;
typedef __attribute__((ext_vector_type(4)))  unsigned int u32x4;
typedef __attribute__((ext_vector_type(8)))  int          i32x8;
typedef __attribute__((ext_vector_type(4)))  int          i32x4;

__device__ __forceinline__ unsigned short f2bf(float f) {
    union { float f; uint32_t u; } c; c.f = f;
    uint32_t r = c.u + 0x7FFFu + ((c.u >> 16) & 1u);   // round-to-nearest-even
    return (unsigned short)(r >> 16);
}

// ---------------------------------------------------------------------------
// Implicit-GEMM conv, v_wmma_f32_16x16x32_bf16, channels-last bf16 input.
//   act : bf16 [N, HW, CinP]   (CinP mult of 64, zero padded)
//   wp  : bf16 [TAPS][Cout][CinP]
//   out : OUT_BF16 ? bf16 [N, HW, CoutP] : f32 [N, Cout, H, W]
// Block: 256 thr (8 waves). Tile 64 Cout x 64 px. Stage K = 64.
// TDMW: weight tile staged by the Tensor Data Mover (TENSOR_LOAD_TO_LDS with
//       pad_interval=32 DWORDs / pad_amount=4 DWORDs -> 144B LDS row pitch),
//       requires Cout % 64 == 0 (fully interior tiles).
// ---------------------------------------------------------------------------
template<int TAPS, bool RELU, bool MASK, bool OUT_BF16, bool TDMW>
__global__ __launch_bounds__(256) void conv_wmma(
    const unsigned short* __restrict__ act,
    const unsigned short* __restrict__ wp,
    const float* __restrict__ bias,
    const int* __restrict__ mask,
    void* __restrict__ outv,
    int CinP, int Cout, int CoutP, int H, int W, int logW)
{
    const int HW    = H * W;
    const int nimg  = blockIdx.y;
    const int pbase = blockIdx.x * 64;
    const int co0   = blockIdx.z * 64;
    const int t     = threadIdx.x;
    const int lane  = t & 31;
    const int wave  = t >> 5;
    const int wx    = wave & 1;        // pixel half (0/1), covers px wx*32..+31
    const int wy    = wave >> 1;       // cout 16-tile (0..3)

    // pitch 72 ushorts = 144B (mult of 16): aligned ds_load_b128 everywhere
    __shared__ __align__(16) unsigned short Xs[64 * 72];   // [px][ci]
    __shared__ __align__(16) unsigned short Ws[64 * 72];   // [co][ci]

    const int lrow = t >> 2;           // 0..63 (tile-load row)
    const int lcol = (t & 3) * 16;     // ushort offset 0/16/32/48

    v8f acc0 = {0.f,0.f,0.f,0.f,0.f,0.f,0.f,0.f};
    v8f acc1 = {0.f,0.f,0.f,0.f,0.f,0.f,0.f,0.f};

    const unsigned short* actn = act + (size_t)nimg * HW * CinP;
    const uint32_t ldsWs = (uint32_t)(uintptr_t)&Ws[0];   // LDS dest offset

    for (int tap = 0; tap < TAPS; ++tap) {
        const int dy = (TAPS == 9) ? (tap / 3 - 1) : 0;
        const int dx = (TAPS == 9) ? (tap % 3 - 1) : 0;
        // hoisted per-tap im2col address for this thread's tile-load row
        const int p  = pbase + lrow;
        const int y  = (p >> logW) + dy;
        const int x  = (p & (W - 1)) + dx;
        const bool vld = ((unsigned)y < (unsigned)H) && ((unsigned)x < (unsigned)W);
        const unsigned short* arow = actn + (size_t)(y * W + x) * CinP + lcol;
        const unsigned short* wrow = (co0 + lrow < Cout)
            ? wp + ((size_t)tap * Cout + co0 + lrow) * CinP + lcol : nullptr;

        for (int ck = 0; ck < CinP; ck += 64) {
            __syncthreads();
            // ---- activation tile: 32B per thread (vector path) -----------
            uint4 a0 = {0,0,0,0}, a1 = {0,0,0,0};
            if (vld) { a0 = *(const uint4*)(arow + ck);
                       a1 = *(const uint4*)(arow + ck + 8); }
            *(uint4*)&Xs[lrow * 72 + lcol]     = a0;
            *(uint4*)&Xs[lrow * 72 + lcol + 8] = a1;
            // ---- weight tile -----------------------------------------------
            if (TDMW) {
                // Tensor Data Mover: 64-row x 128B tile, 16B LDS pad per row
                if (wave == 0) {
                    const uint64_t ga = (uint64_t)(uintptr_t)wp +
                        (((size_t)tap * Cout + co0) * CinP + ck) * 2;
                    u32x4 g0;
                    g0[0] = 1u;                                   // count=1 (valid D#)
                    g0[1] = ldsWs;                                // lds_addr
                    g0[2] = (uint32_t)ga;                         // global_addr lo
                    g0[3] = (uint32_t)(ga >> 32) | (2u << 30);    // addr hi | type=image
                    i32x8 g1;
                    g1[0] = (int)((1u << 16)    // data_size = 2 bytes
                                | (1u << 20)    // pad_enable
                                | (4u << 22)    // pad_interval: 32 DWORDs (128B)
                                | (3u << 25));  // pad_amount:   4 DWORDs (16B)
                    g1[1] = (int)(((unsigned)CinP & 0xFFFFu) << 16);        // tensor_dim0 lo
                    g1[2] = (int)(((unsigned)CinP >> 16)
                                | (((unsigned)Cout & 0xFFFFu) << 16));      // dim0 hi | dim1 lo
                    g1[3] = (int)(((unsigned)Cout >> 16) | (64u << 16));    // dim1 hi | tile_dim0=64
                    g1[4] = 64;                                             // tile_dim1=64
                    g1[5] = (int)CinP;                                      // dim0_stride lo
                    g1[6] = 0;                                              // stride hi
                    g1[7] = 0;
                    i32x4 gz4 = {0, 0, 0, 0};                               // 2D: groups 2/3 NULL
                    i32x8 gz8 = {0, 0, 0, 0, 0, 0, 0, 0};
                    __builtin_amdgcn_tensor_load_to_lds(g0, g1, gz4, gz4, gz8, 0);
                    __builtin_amdgcn_s_wait_tensorcnt(0);
                    asm volatile("" ::: "memory");
                }
            } else {
                uint4 w0 = {0,0,0,0}, w1 = {0,0,0,0};
                if (wrow) { w0 = *(const uint4*)(wrow + ck);
                            w1 = *(const uint4*)(wrow + ck + 8);
                            if (ck + 64 < CinP)
                                __builtin_prefetch(wrow + ck + 64, 0, 1); }
                *(uint4*)&Ws[lrow * 72 + lcol]     = w0;
                *(uint4*)&Ws[lrow * 72 + lcol + 8] = w1;
            }
            __syncthreads();
            // ---- 2 K-halves x 2 pixel tiles = 4 WMMA ---------------------
            const int cr = lane & 15;
            const int ka = (lane < 16) ? 0 : 16;   // A lanes: K0-15 vs K16-31
            const int kb = (lane < 16) ? 0 : 16;   // B lanes: K0-15 vs K16-31
            const int px0 = (lane & 15) + wx * 32;
            #pragma unroll
            for (int kk = 0; kk < 64; kk += 32) {
                union { uint4 u[2]; v16bf v; } A, B0, B1;
                const unsigned short* wr = &Ws[(wy * 16 + cr) * 72 + kk + ka];
                A.u[0]  = *(const uint4*)(wr);
                A.u[1]  = *(const uint4*)(wr + 8);
                const unsigned short* x0 = &Xs[px0 * 72 + kk + kb];
                B0.u[0] = *(const uint4*)(x0);
                B0.u[1] = *(const uint4*)(x0 + 8);
                const unsigned short* x1 = &Xs[(px0 + 16) * 72 + kk + kb];
                B1.u[0] = *(const uint4*)(x1);
                B1.u[1] = *(const uint4*)(x1 + 8);
                acc0 = __builtin_amdgcn_wmma_f32_16x16x32_bf16(
                           false, A.v, false, B0.v, (short)0, acc0, false, false);
                acc1 = __builtin_amdgcn_wmma_f32_16x16x32_bf16(
                           false, A.v, false, B1.v, (short)0, acc1, false, false);
            }
        }
    }

    // ---- epilogue: bias + relu + mask ------------------------------------
    const int hi8    = (lane >= 16) ? 8 : 0;   // D: lanes 16-31 hold M = g+8
    const int cobase = co0 + wy * 16 + hi8;
    const int pxA    = (lane & 15) + wx * 32;
    #pragma unroll
    for (int half = 0; half < 2; ++half) {
        const v8f& acc = half ? acc1 : acc0;
        const int px = pxA + half * 16;
        const int pp = pbase + px;
        float mv = 1.f;
        if (MASK) mv = (mask[(size_t)nimg * HW + pp] > 0) ? 1.f : 0.f;
        if (OUT_BF16) {
            // NHWC bf16: 8 consecutive couts per lane -> single b128 store
            union { unsigned short s[8]; uint4 u; } ob;
            #pragma unroll
            for (int g = 0; g < 8; ++g) {
                float v = acc[g] + bias[cobase + g];
                if (RELU) v = fmaxf(v, 0.f);
                ob.s[g] = f2bf(v * mv);
            }
            *(uint4*)((unsigned short*)outv +
                      ((size_t)nimg * HW + pp) * CoutP + cobase) = ob.u;
        } else {
            float* out = (float*)outv;
            #pragma unroll
            for (int g = 0; g < 8; ++g) {
                const int co = cobase + g;
                if (co < Cout) {
                    float v = acc[g] + bias[co];
                    if (RELU) v = fmaxf(v, 0.f);
                    out[((size_t)nimg * Cout + co) * HW + pp] = v * mv;
                }
            }
        }
    }
}

// ---------------------------------------------------------------------------
// NCHW f32 -> NHWC bf16 (padded CinP), tiled LDS transpose, coalesced both ways
// grid: (HW/64, CinP/64, N), 256 threads
// ---------------------------------------------------------------------------
__global__ __launch_bounds__(256) void cvt_nhwc(
    const float* __restrict__ in, unsigned short* __restrict__ out,
    int C, int CinP, int HW)
{
    __shared__ __align__(16) unsigned short T[64 * 72];
    const int t = threadIdx.x;
    const int pbase = blockIdx.x * 64;
    const int cbase = blockIdx.y * 64;
    const int n = blockIdx.z;
    {
        const int ci = cbase + (t >> 2);
        const int pxc = (t & 3) * 16;
        if (ci < C) {
            const float* src = in + ((size_t)n * C + ci) * HW + pbase + pxc;
            #pragma unroll
            for (int j = 0; j < 16; ++j)
                T[(pxc + j) * 72 + (t >> 2)] = f2bf(src[j]);
        } else {
            #pragma unroll
            for (int j = 0; j < 16; ++j)
                T[(pxc + j) * 72 + (t >> 2)] = 0;
        }
    }
    __syncthreads();
    {
        const int px = t >> 2;
        const int cc = (t & 3) * 16;
        unsigned short* dst = out + ((size_t)n * HW + pbase + px) * CinP + cbase + cc;
        *(uint4*)(dst)     = *(uint4*)&T[px * 72 + cc];
        *(uint4*)(dst + 8) = *(uint4*)&T[px * 72 + cc + 8];
    }
}

// ---------------------------------------------------------------------------
// Concat [x(256)|rel(2)|abs(2)|pad->320] -> NHWC bf16 [N,HW,320]
// grid: (HW/64, 5, N)
// ---------------------------------------------------------------------------
__global__ __launch_bounds__(256) void concat_nhwc(
    const float* __restrict__ x, const float* __restrict__ rel,
    const float* __restrict__ abs_, unsigned short* __restrict__ out, int HW)
{
    __shared__ __align__(16) unsigned short T[64 * 72];
    const int t = threadIdx.x;
    const int pbase = blockIdx.x * 64;
    const int cbase = blockIdx.y * 64;
    const int n = blockIdx.z;
    {
        const int ci = cbase + (t >> 2);
        const int pxc = (t & 3) * 16;
        const float* src = nullptr;
        if (ci < 256)      src = x    + ((size_t)n * 256 + ci)         * HW;
        else if (ci < 258) src = rel  + ((size_t)n * 2 + (ci - 256))   * HW;
        else if (ci < 260) src = abs_ + ((size_t)n * 2 + (ci - 258))   * HW;
        if (src) {
            src += pbase + pxc;
            #pragma unroll
            for (int j = 0; j < 16; ++j)
                T[(pxc + j) * 72 + (t >> 2)] = f2bf(src[j]);
        } else {
            #pragma unroll
            for (int j = 0; j < 16; ++j)
                T[(pxc + j) * 72 + (t >> 2)] = 0;
        }
    }
    __syncthreads();
    {
        const int px = t >> 2;
        const int cc = (t & 3) * 16;
        unsigned short* dst = out + ((size_t)n * HW + pbase + px) * 320 + cbase + cc;
        *(uint4*)(dst)     = *(uint4*)&T[px * 72 + cc];
        *(uint4*)(dst + 8) = *(uint4*)&T[px * 72 + cc + 8];
    }
}

// ---------------------------------------------------------------------------
// Bilinear 2x upsample (align_corners=False, clamped), optional accumulate.
// ---------------------------------------------------------------------------
template<bool ADD>
__global__ void upsample2x(const float* __restrict__ in, float* __restrict__ out,
                           int N, int C, int H, int W)
{
    const int H2 = H * 2, W2 = W * 2;
    const size_t total = (size_t)N * C * H2 * W2;
    const size_t i = (size_t)blockIdx.x * blockDim.x + threadIdx.x;
    if (i >= total) return;
    const int ox = (int)(i % W2);
    size_t r = i / W2;
    const int oy = (int)(r % H2);
    const size_t ncc = r / H2;
    const float sy = oy * 0.5f - 0.25f;
    const float sx = ox * 0.5f - 0.25f;
    int y0 = (int)floorf(sy); const float fy = sy - y0;
    int x0 = (int)floorf(sx); const float fx = sx - x0;
    int y1 = y0 + 1, x1 = x0 + 1;
    y0 = min(max(y0, 0), H - 1); y1 = min(max(y1, 0), H - 1);
    x0 = min(max(x0, 0), W - 1); x1 = min(max(x1, 0), W - 1);
    const float* ip = in + ncc * (size_t)H * W;
    const float v = (1.f - fy) * ((1.f - fx) * ip[y0 * W + x0] + fx * ip[y0 * W + x1])
                  +        fy  * ((1.f - fx) * ip[y1 * W + x0] + fx * ip[y1 * W + x1]);
    if (ADD) out[i] += v; else out[i] = v;
}

// ---------------------------------------------------------------------------
// Weight pack: f32 OIHW -> bf16 [KK][Cout][CinP], zero pad
// ---------------------------------------------------------------------------
__global__ void pack_w(const float* __restrict__ src, unsigned short* __restrict__ dst,
                       int Cout, int Cin, int CinP, int KK)
{
    const size_t total = (size_t)KK * Cout * CinP;
    const size_t i = (size_t)blockIdx.x * blockDim.x + threadIdx.x;
    if (i >= total) return;
    const int ci = (int)(i % CinP);
    size_t r = i / CinP;
    const int co  = (int)(r % Cout);
    const int tap = (int)(r / Cout);
    float v = 0.f;
    if (ci < Cin) v = src[((size_t)co * Cin + ci) * KK + tap];
    dst[i] = f2bf(v);
}

// ---------------------------------------------------------------------------
extern "C" void kernel_launch(void* const* d_in, const int* in_sizes, int n_in,
                              void* d_out, int out_size, void* d_ws, size_t ws_size,
                              hipStream_t stream)
{
    const float* p2   = (const float*)d_in[0];
    const float* p3   = (const float*)d_in[1];
    const float* p4   = (const float*)d_in[2];
    const float* p5   = (const float*)d_in[3];
    const float* rel  = (const float*)d_in[4];
    const float* abs_ = (const float*)d_in[5];
    const int*   fg   = (const int*)  d_in[6];
    const float* w_p2_0 = (const float*)d_in[7];  const float* b_p2_0 = (const float*)d_in[8];
    const float* w_p3_0 = (const float*)d_in[9];  const float* b_p3_0 = (const float*)d_in[10];
    const float* w_p4_0 = (const float*)d_in[11]; const float* b_p4_0 = (const float*)d_in[12];
    const float* w_p4_1 = (const float*)d_in[13]; const float* b_p4_1 = (const float*)d_in[14];
    const float* w_p5_0 = (const float*)d_in[15]; const float* b_p5_0 = (const float*)d_in[16];
    const float* w_p5_1 = (const float*)d_in[17]; const float* b_p5_1 = (const float*)d_in[18];
    const float* w_p5_2 = (const float*)d_in[19]; const float* b_p5_2 = (const float*)d_in[20];
    const float* comb_w = (const float*)d_in[21]; const float* comb_b = (const float*)d_in[22];
    const float* head_w0= (const float*)d_in[23]; const float* head_b0= (const float*)d_in[24];
    const float* head_w = (const float*)d_in[25]; const float* head_b = (const float*)d_in[26];
    const float* pred_w = (const float*)d_in[27]; const float* pred_b = (const float*)d_in[28];

    const int N = 4, C = 256, D = 512, NCLS = 75;
    const int HW = 128 * 128;

    char* ws = (char*)d_ws;
    size_t off = 0;
    auto alloc = [&](size_t b) -> char* {
        char* pp = ws + off; off += (b + 255) & ~(size_t)255; return pp;
    };
    float* X    = (float*)alloc((size_t)N * C * HW    * 4);
    float* T64A = (float*)alloc((size_t)N * C * 64*64 * 4);
    float* T64B = (float*)alloc((size_t)N * C * 64*64 * 4);
    float* T32A = (float*)alloc((size_t)N * C * 32*32 * 4);
    float* T32B = (float*)alloc((size_t)N * C * 32*32 * 4);
    float* T16  = (float*)alloc((size_t)N * C * 16*16 * 4);
    typedef unsigned short us;
    us* A128 = (us*)alloc((size_t)N * HW    * 256 * 2);
    us* A64  = (us*)alloc((size_t)N * 64*64 * 256 * 2);
    us* A32  = (us*)alloc((size_t)N * 32*32 * 256 * 2);
    us* A16  = (us*)alloc((size_t)N * 16*16 * 256 * 2);
    us* A260 = (us*)alloc((size_t)N * HW    * 320 * 2);
    us* Hn1  = (us*)alloc((size_t)N * HW    * 512 * 2);
    us* Hn2  = (us*)alloc((size_t)N * HW    * 512 * 2);
    us* pw_p2  = (us*)alloc((size_t)9 * C * 256 * 2);
    us* pw_p3  = (us*)alloc((size_t)9 * C * 256 * 2);
    us* pw_p4a = (us*)alloc((size_t)9 * C * 256 * 2);
    us* pw_p4b = (us*)alloc((size_t)9 * C * 256 * 2);
    us* pw_p5a = (us*)alloc((size_t)9 * C * 256 * 2);
    us* pw_p5b = (us*)alloc((size_t)9 * C * 256 * 2);
    us* pw_p5c = (us*)alloc((size_t)9 * C * 256 * 2);
    us* pw_comb= (us*)alloc((size_t)9 * C * 320 * 2);
    us* pw_h0  = (us*)alloc((size_t)9 * D * 256 * 2);
    us* pw_h[7];
    for (int l = 0; l < 7; ++l) pw_h[l] = (us*)alloc((size_t)9 * D * D * 2);
    us* pw_pred= (us*)alloc((size_t)1 * NCLS * D * 2);

    auto pack = [&](const float* src, us* dst, int Co, int Ci, int CiP, int KK) {
        size_t total = (size_t)KK * Co * CiP;
        pack_w<<<(unsigned)((total + 255) / 256), 256, 0, stream>>>(src, dst, Co, Ci, CiP, KK);
    };
    auto cvt = [&](const float* src, us* dst, int Ci, int CiP, int hw) {
        dim3 g(hw / 64, CiP / 64, N);
        cvt_nhwc<<<g, 256, 0, stream>>>(src, dst, Ci, CiP, hw);
    };
    auto convF = [&](const us* in, const us* wpk, const float* bias,
                     float* outp, int CiP, int Co, int Hh, int Ww) {
        dim3 g((Hh * Ww) / 64, N, Co / 64);
        conv_wmma<9, true, false, false, true><<<g, 256, 0, stream>>>(
            in, wpk, bias, nullptr, outp, CiP, Co, 0, Hh, Ww, __builtin_ctz(Ww));
    };
    auto convB = [&](const us* in, const us* wpk, const float* bias,
                     us* outp, int CiP, int Co) {
        dim3 g(HW / 64, N, Co / 64);
        conv_wmma<9, true, true, true, true><<<g, 256, 0, stream>>>(
            in, wpk, bias, fg, outp, CiP, Co, Co, 128, 128, 7);
    };
    auto up = [&](const float* in, float* outp, int Hh, int Ww, bool add) {
        size_t total = (size_t)N * C * (2 * Hh) * (2 * Ww);
        unsigned g = (unsigned)((total + 255) / 256);
        if (add) upsample2x<true ><<<g, 256, 0, stream>>>(in, outp, N, C, Hh, Ww);
        else     upsample2x<false><<<g, 256, 0, stream>>>(in, outp, N, C, Hh, Ww);
    };

    // ---- pack weights ------------------------------------------------------
    pack(w_p2_0, pw_p2,  C, C, 256, 9);
    pack(w_p3_0, pw_p3,  C, C, 256, 9);
    pack(w_p4_0, pw_p4a, C, C, 256, 9);
    pack(w_p4_1, pw_p4b, C, C, 256, 9);
    pack(w_p5_0, pw_p5a, C, C, 256, 9);
    pack(w_p5_1, pw_p5b, C, C, 256, 9);
    pack(w_p5_2, pw_p5c, C, C, 256, 9);
    pack(comb_w, pw_comb, C, 260, 320, 9);
    pack(head_w0, pw_h0, D, C, 256, 9);
    for (int l = 0; l < 7; ++l)
        pack(head_w + (size_t)l * D * D * 9, pw_h[l], D, D, D, 9);
    pack(pred_w, pw_pred, NCLS, D, D, 1);

    // ---- FPN scale heads (f32 NCHW side) -----------------------------------
    cvt(p2, A128, C, 256, HW);
    convF(A128, pw_p2, b_p2_0, X, 256, C, 128, 128);
    cvt(p3, A64, C, 256, 64*64);
    convF(A64, pw_p3, b_p3_0, T64A, 256, C, 64, 64);
    up(T64A, X, 64, 64, true);
    cvt(p4, A32, C, 256, 32*32);
    convF(A32, pw_p4a, b_p4_0, T32A, 256, C, 32, 32);
    up(T32A, T64A, 32, 32, false);
    cvt(T64A, A64, C, 256, 64*64);
    convF(A64, pw_p4b, b_p4_1, T64B, 256, C, 64, 64);
    up(T64B, X, 64, 64, true);
    cvt(p5, A16, C, 256, 16*16);
    convF(A16, pw_p5a, b_p5_0, T16, 256, C, 16, 16);
    up(T16, T32A, 16, 16, false);
    cvt(T32A, A32, C, 256, 32*32);
    convF(A32, pw_p5b, b_p5_1, T32B, 256, C, 32, 32);
    up(T32B, T64A, 32, 32, false);
    cvt(T64A, A64, C, 256, 64*64);
    convF(A64, pw_p5c, b_p5_2, T64B, 256, C, 64, 64);
    up(T64B, X, 64, 64, true);

    // ---- PE concat -> comb conv (masked output == h0, NHWC bf16) -----------
    {
        dim3 g(HW / 64, 5, N);
        concat_nhwc<<<g, 256, 0, stream>>>(X, rel, abs_, A260, HW);
    }
    convB(A260, pw_comb, comb_b, A128, 320, C);   // A128 := masked h0 [N,HW,256]

    // ---- sparse head: 8 masked convs on NHWC bf16 fast path ----------------
    convB(A128, pw_h0, head_b0, Hn1, 256, D);
    us* src = Hn1; us* dst = Hn2;
    for (int l = 0; l < 7; ++l) {
        convB(src, pw_h[l], head_b + (size_t)l * D, dst, D, D);
        us* tmp = src; src = dst; dst = tmp;
    }

    // ---- predictor: 1x1 conv 512 -> 75, f32 NCHW to d_out (vector weights) -
    {
        dim3 g(HW / 64, N, (NCLS + 63) / 64);
        conv_wmma<1, false, false, false, false><<<g, 256, 0, stream>>>(
            src, pw_pred, pred_b, nullptr, d_out, D, NCLS, 0, 128, 128, 7);
    }
}
